// RelNet_47828755808730
// MI455X (gfx1250) — matrix operand order repaired and checked by reading, not compiled, not run on hardware
//
#include <hip/hip_runtime.h>
#include <hip/hip_bf16.h>

// ---------------------------------------------------------------------------
// RelNet fused kernel for gfx1250 (MI455X), wave32 + WMMA f16.
//   k1: hi/hj projection (8x256x128 each) + zero pooled accumulator
//   k2: fused pair-GEMM  g = relu(relu(hi[i]+hj[j]+bg1) @ Wg2 + bg2),
//       pooled += sum_{i,j} g   -- never materializes the 268MB hpair tensor.
//       v_wmma_f32_16x16x32_f16; A built on the fly (bg1 pre-folded into the
//       staged hi tile), B pre-swizzled f16 Wg2 resident in LDS.
//   k3: pooled -> MLP -> broadcast to (8,64,16,16)
// ---------------------------------------------------------------------------

typedef _Float16 v16h __attribute__((ext_vector_type(16)));
typedef float    v8f  __attribute__((ext_vector_type(8)));

#define B_   8
#define C_   64
#define N_   256   // H*W
#define GH_  128
#define GO_  64
#define FH_  128
#define FO_  64

// workspace layout (in floats)
#define WS_HI   0
#define WS_HJ   (B_*N_*GH_)        // 262144
#define WS_POOL (2*B_*N_*GH_)      // 524288 (+512)

union AFrag { v16h v; _Float16 h[16]; };
union CFrag { v8f  v; float    f[8];  };

// ------------------------------ kernel 1 -----------------------------------
__global__ void __launch_bounds__(256)
k1_proj(const float* __restrict__ x, const float* __restrict__ Wg1,
        float* __restrict__ hi, float* __restrict__ hj,
        float* __restrict__ pooled)
{
    const int tid = blockIdx.x * 256 + threadIdx.x;      // 0 .. 262143
    if (tid < B_ * GO_) pooled[tid] = 0.0f;              // zero pooled accum

    const int g = tid & (GH_ - 1);
    const int n = (tid >> 7) & (N_ - 1);
    const int b = tid >> 15;

    const float* xp = x + (size_t)(b * C_) * N_ + n;     // x[b][c][n], stride N_
    float ai = 0.0f, aj = 0.0f;
    #pragma unroll 4
    for (int c = 0; c < C_; ++c) {
        const float xv = xp[c * N_];
        ai = fmaf(xv, Wg1[c * GH_ + g],        ai);      // Wi = Wg1[:64]
        aj = fmaf(xv, Wg1[(C_ + c) * GH_ + g], aj);      // Wj = Wg1[64:]
    }
    hi[tid] = ai;
    hj[tid] = aj;
}

// ------------------------------ kernel 2 -----------------------------------
// grid = B_ * 16 blocks; block = (b, i-tile of 16 i's); 8 waves per block,
// each wave handles j-tiles {wv, wv+8}, 16 i's per j-tile -> 32 M-tiles/wave.
__global__ void __launch_bounds__(256)
k2_pair(const float* __restrict__ hi, const float* __restrict__ hj,
        const float* __restrict__ bg1, const float* __restrict__ Wg2,
        const float* __restrict__ bg2, float* __restrict__ pooled)
{
    __shared__ float    hiT[16 * GH_];          // 16 i-rows of hi+bg1   (8 KB)
    __shared__ _Float16 ldsB[4 * 4 * 32 * 16];  // swizzled f16 Wg2     (16 KB)
    __shared__ float    bg2L[GO_];

    const int b  = blockIdx.x >> 4;
    const int i0 = (blockIdx.x & 15) << 4;
    const int t  = threadIdx.x;

    // ---- stage hi tile with bg1 pre-folded (loop-invariant add hoisted) ----
    for (int e = t; e < 16 * GH_; e += 256)
        hiT[e] = hi[(size_t)(b * N_ + i0) * GH_ + e] + bg1[e & (GH_ - 1)];
    for (int e = t; e < GO_; e += 256) bg2L[e] = bg2[e];

    // ---- swizzle Wg2 (128x64 f32) into per-lane B fragments (f16) ----
    // fragment index: (((ks*4 + nt)*32 + lane)*16 + h)
    // B layout (16x16x32 f16): n = nt*16 + (lane&15);
    //                          k = ks*32 + (lane>=16 ? 8 : 0) + h + (h>=8 ? 8 : 0)
    for (int e = t; e < 4 * 4 * 32 * 16; e += 256) {
        const int h    = e & 15;
        const int lane = (e >> 4) & 31;
        const int nt   = (e >> 9) & 3;
        const int ks   = (e >> 11) & 3;
        const int k    = ks * 32 + ((lane >= 16) ? 8 : 0) + h + ((h >= 8) ? 8 : 0);
        const int n    = nt * 16 + (lane & 15);
        ldsB[e] = (_Float16)Wg2[k * GO_ + n];
    }
    __syncthreads();

    const int wv   = t >> 5;
    const int lane = t & 31;
    const int m    = lane & 15;                 // row within M-tile / N column
    const int loff = (lane >= 16) ? 8 : 0;      // A-layout K offset for hi lanes

    float poolAcc[4] = {0.f, 0.f, 0.f, 0.f};

    for (int jrep = 0; jrep < 2; ++jrep) {
        const int jt = wv + jrep * 8;
        const int j0 = jt << 4;
        // this lane's hj row: pair row m of the M-tile is j = j0 + m
        const float* __restrict__ hjRow = hj + (size_t)(b * N_ + j0 + m) * GH_;

        for (int il = 0; il < 16; ++il) {
            const float* __restrict__ hiRow = hiT + il * GH_;

            CFrag c0{}, c1{}, c2{}, c3{};

            #pragma unroll
            for (int ks = 0; ks < 4; ++ks) {
                // ---- build A fragment: relu((hi+bg1) + hj) -> f16 ----
                // A layout (16x32 f16): lane holds row m; K(h) below per ISA.
                AFrag a;
                const int kb = ks * 32 + loff;
                #pragma unroll
                for (int h = 0; h < 16; ++h) {
                    const int K = kb + h + ((h >= 8) ? 8 : 0);
                    const float v = hiRow[K] + hjRow[K];
                    a.h[h] = (_Float16)fmaxf(v, 0.0f);
                }

                const v16h* __restrict__ bBase =
                    (const v16h*)ldsB + (ks * 4) * 32 + lane;

                c0.v = __builtin_amdgcn_wmma_f32_16x16x32_f16(
                          false, a.v, false, bBase[0 * 32], (short)0, c0.v, false, false);
                c1.v = __builtin_amdgcn_wmma_f32_16x16x32_f16(
                          false, a.v, false, bBase[1 * 32], (short)0, c1.v, false, false);
                c2.v = __builtin_amdgcn_wmma_f32_16x16x32_f16(
                          false, a.v, false, bBase[2 * 32], (short)0, c2.v, false, false);
                c3.v = __builtin_amdgcn_wmma_f32_16x16x32_f16(
                          false, a.v, false, bBase[3 * 32], (short)0, c3.v, false, false);
            }

            // ---- epilogue: relu(c + bg2) then reduce over M into poolAcc ----
            // C/D layout: N = lane&15, M = r + 8*(lane>=16); summing over M
            // means every element of each accumulator feeds column n.
            #pragma unroll
            for (int r = 0; r < 8; ++r) {
                poolAcc[0] += fmaxf(c0.f[r] + bg2L[ 0 + m], 0.0f);
                poolAcc[1] += fmaxf(c1.f[r] + bg2L[16 + m], 0.0f);
                poolAcc[2] += fmaxf(c2.f[r] + bg2L[32 + m], 0.0f);
                poolAcc[3] += fmaxf(c3.f[r] + bg2L[48 + m], 0.0f);
            }
        }
    }

    #pragma unroll
    for (int nt = 0; nt < 4; ++nt)
        atomicAdd(&pooled[b * GO_ + nt * 16 + m], poolAcc[nt]);
}

// ------------------------------ kernel 3 -----------------------------------
__global__ void __launch_bounds__(256)
k3_mlp(const float* __restrict__ pooled,
       const float* __restrict__ Wf1, const float* __restrict__ bf1,
       const float* __restrict__ Wf2, const float* __restrict__ bf2,
       float* __restrict__ out)
{
    __shared__ float pL[B_ * GO_];    // 512
    __shared__ float f1L[B_ * FH_];   // 1024
    const int t = threadIdx.x;

    for (int e = t; e < B_ * GO_; e += 256) pL[e] = pooled[e];
    __syncthreads();

    for (int e = t; e < B_ * FH_; e += 256) {
        const int b = e >> 7, fh = e & 127;
        float acc = bf1[fh];
        #pragma unroll 4
        for (int c = 0; c < GO_; ++c)
            acc = fmaf(pL[b * GO_ + c], Wf1[c * FH_ + fh], acc);
        f1L[e] = fmaxf(acc, 0.0f);
    }
    __syncthreads();

    for (int e = t; e < B_ * FO_; e += 256) {
        const int b = e >> 6, fo = e & 63;
        float acc = bf2[fo];
        #pragma unroll 4
        for (int k = 0; k < FH_; ++k)
            acc = fmaf(f1L[b * FH_ + k], Wf2[k * FO_ + fo], acc);
        // broadcast over 16x16 spatial positions: out[b][fo][:][:]
        float* __restrict__ op = out + (size_t)e * N_;
        #pragma unroll 4
        for (int p = 0; p < N_; ++p) op[p] = acc;
    }
}

// ------------------------------ launch -------------------------------------
extern "C" void kernel_launch(void* const* d_in, const int* in_sizes, int n_in,
                              void* d_out, int out_size, void* d_ws, size_t ws_size,
                              hipStream_t stream)
{
    const float* x   = (const float*)d_in[0];
    const float* Wg1 = (const float*)d_in[1];
    const float* bg1 = (const float*)d_in[2];
    const float* Wg2 = (const float*)d_in[3];
    const float* bg2 = (const float*)d_in[4];
    const float* Wf1 = (const float*)d_in[5];
    const float* bf1 = (const float*)d_in[6];
    const float* Wf2 = (const float*)d_in[7];
    const float* bf2 = (const float*)d_in[8];
    float*       out = (float*)d_out;

    float* ws     = (float*)d_ws;
    float* hi     = ws + WS_HI;
    float* hj     = ws + WS_HJ;
    float* pooled = ws + WS_POOL;

    // k1: 8*256*128 threads, one (b,n,g) each
    k1_proj<<<(B_ * N_ * GH_) / 256, 256, 0, stream>>>(x, Wg1, hi, hj, pooled);
    // k2: one block per (b, i-tile)
    k2_pair<<<B_ * 16, 256, 0, stream>>>(hi, hj, bg1, Wg2, bg2, pooled);
    // k3: single block finishes the MLP + broadcast
    k3_mlp<<<1, 256, 0, stream>>>(pooled, Wf1, bf1, Wf2, bf2, out);
}
